// TTTInner_12223476924503
// MI455X (gfx1250) — compile-verified, matrix operand order checked by reading, not run on hardware
//
#include <hip/hip_runtime.h>

// ---------------------------------------------------------------------------
// TTT inner step on gfx1250: all 6 GEMMs via v_wmma_f32_16x16x32_bf16.
// proj: 64x64 block tile (3 fused accumulator sets). Generic GEMM: 64x128
// block tile, 4 waves x (32x64) = 2x4 WMMA frags/wave. All global staging
// vectorized to 16B loads; fp32->bf16 converted inline while staging to LDS.
// ---------------------------------------------------------------------------

typedef __attribute__((ext_vector_type(16))) __bf16 v16bf;
typedef __attribute__((ext_vector_type(8)))  float  v8f;

constexpr int Dd = 2048;   // feature dim
constexpr int Bt = 4096;   // tokens
constexpr int TK  = 32;    // K-step
constexpr int LDK = 40;    // padded halfword row stride (80B: 16B-aligned, bank-spread)
constexpr int NT  = 128;   // 4 waves per block

constexpr int TMp = 64, TNp = 64;     // proj block tile
constexpr int TMg = 64, TNg = 128;    // generic GEMM block tile

__device__ __forceinline__ unsigned short f2bf(float f) {
  unsigned u = __float_as_uint(f);
  u += 0x7FFFu + ((u >> 16) & 1u);           // round-to-nearest-even
  return (unsigned short)(u >> 16);
}
__device__ __forceinline__ float bf2f(unsigned short h) {
  return __uint_as_float(((unsigned)h) << 16);
}
__device__ __forceinline__ unsigned pk2(float a, float b) {
  return (unsigned)f2bf(a) | ((unsigned)f2bf(b) << 16);
}

// Load one 16x32 bf16 WMMA fragment for this lane from an LDS tile stored
// row-major (row = M for A-tiles, row = N for B-tiles), K contiguous.
// Lane layout (ISA 7.12.2): m/n = lane%16, half = lane/16 picks K-groups
// {half*8 .. +7} and {16+half*8 .. +7}.
__device__ __forceinline__ v16bf ldfrag(const unsigned short* row, int half) {
  union { uint4 q[2]; v16bf v; } u;
  u.q[0] = *reinterpret_cast<const uint4*>(row + half * 8);
  u.q[1] = *reinterpret_cast<const uint4*>(row + 16 + half * 8);
  return u.v;
}

// ---------------------------------------------------------------------------
// Kernel 1: fused triple projection. train=src@thk, test=src@thq,
// target=(src@thv)-train. Shares A fragments across 3 accumulator sets.
// ---------------------------------------------------------------------------
__global__ __launch_bounds__(NT)
void proj_kernel(const float* __restrict__ src,
                 const float* __restrict__ thk,
                 const float* __restrict__ thq,
                 const float* __restrict__ thv,
                 unsigned short* __restrict__ trainb,
                 unsigned short* __restrict__ testb,
                 float* __restrict__ testf,
                 unsigned short* __restrict__ targetb)
{
  __shared__ __align__(16) unsigned short As[TMp * LDK];
  __shared__ __align__(16) unsigned short Bk[TNp * LDK];
  __shared__ __align__(16) unsigned short Bq[TNp * LDK];
  __shared__ __align__(16) unsigned short Bv[TNp * LDK];

  const int tid = threadIdx.x, wave = tid >> 5, lane = tid & 31;
  const int lh = lane >> 4, li = lane & 15;
  const int wm = (wave >> 1) * 32, wn = (wave & 1) * 32;
  const int m0 = blockIdx.y * TMp, n0 = blockIdx.x * TNp;

  const v8f vz = {0.f,0.f,0.f,0.f,0.f,0.f,0.f,0.f};
  v8f ak[2][2], aq[2][2], av[2][2];
  for (int i = 0; i < 2; i++)
    for (int j = 0; j < 2; j++) { ak[i][j] = vz; aq[i][j] = vz; av[i][j] = vz; }

  for (int k0 = 0; k0 < Dd; k0 += TK) {
    // A: src[m][k] f32, k-contiguous: float4 load -> packed uint2 LDS store
    for (int q = tid; q < (TMp * TK) / 4; q += NT) {   // 512 chunks of 4
      int m = q >> 3, kc = (q & 7) * 4;
      float4 f = *reinterpret_cast<const float4*>(src + (size_t)(m0 + m) * Dd + (k0 + kc));
      uint2 r; r.x = pk2(f.x, f.y); r.y = pk2(f.z, f.w);
      *reinterpret_cast<uint2*>(&As[m * LDK + kc]) = r;
    }
    // B: theta[k][n] f32, n-contiguous: float4 loads, scatter-stores to Bs[n][k]
    for (int q = tid; q < (TNp * TK) / 4; q += NT) {   // 512 chunks of 4
      int k = q >> 4, nc = (q & 15) * 4;
      size_t g = (size_t)(k0 + k) * Dd + (n0 + nc);
      float4 fk = *reinterpret_cast<const float4*>(thk + g);
      float4 fq = *reinterpret_cast<const float4*>(thq + g);
      float4 fv = *reinterpret_cast<const float4*>(thv + g);
      int s = nc * LDK + k;
      Bk[s          ] = f2bf(fk.x); Bk[s + LDK    ] = f2bf(fk.y);
      Bk[s + 2 * LDK] = f2bf(fk.z); Bk[s + 3 * LDK] = f2bf(fk.w);
      Bq[s          ] = f2bf(fq.x); Bq[s + LDK    ] = f2bf(fq.y);
      Bq[s + 2 * LDK] = f2bf(fq.z); Bq[s + 3 * LDK] = f2bf(fq.w);
      Bv[s          ] = f2bf(fv.x); Bv[s + LDK    ] = f2bf(fv.y);
      Bv[s + 2 * LDK] = f2bf(fv.z); Bv[s + 3 * LDK] = f2bf(fv.w);
    }
    __syncthreads();
    v16bf af[2];
    af[0] = ldfrag(&As[(wm + li) * LDK], lh);
    af[1] = ldfrag(&As[(wm + 16 + li) * LDK], lh);
    for (int nf = 0; nf < 2; nf++) {
      v16bf bk = ldfrag(&Bk[(wn + nf * 16 + li) * LDK], lh);
      v16bf bq = ldfrag(&Bq[(wn + nf * 16 + li) * LDK], lh);
      v16bf bv = ldfrag(&Bv[(wn + nf * 16 + li) * LDK], lh);
      for (int mf = 0; mf < 2; mf++) {
        ak[mf][nf] = __builtin_amdgcn_wmma_f32_16x16x32_bf16(false, af[mf], false, bk, (short)0, ak[mf][nf], false, false);
        aq[mf][nf] = __builtin_amdgcn_wmma_f32_16x16x32_bf16(false, af[mf], false, bq, (short)0, aq[mf][nf], false, false);
        av[mf][nf] = __builtin_amdgcn_wmma_f32_16x16x32_bf16(false, af[mf], false, bv, (short)0, av[mf][nf], false, false);
      }
    }
    __syncthreads();
  }

  for (int mf = 0; mf < 2; mf++)
    for (int nf = 0; nf < 2; nf++)
      for (int r = 0; r < 8; r++) {
        int row = m0 + wm + mf * 16 + lh * 8 + r;
        int col = n0 + wn + nf * 16 + li;
        size_t o = (size_t)row * Dd + col;
        float tk = ak[mf][nf][r];
        float tq = aq[mf][nf][r];
        float tv = av[mf][nf][r];
        trainb[o]  = f2bf(tk);
        testb[o]   = f2bf(tq);
        testf[o]   = tq;
        targetb[o] = f2bf(tv - tk);
      }
}

// ---------------------------------------------------------------------------
// Generic WMMA GEMM (64x128 block tile, 2x4 frags/wave) with fused epilogues.
// AM:  0 = bf16 A[M,K] row-major      1 = bf16 A[m][k]=Ag[k*lda+m] (err^T)
// BMO: 1 = f32  B[k][n]=Bg[n*ldb+k]   2 = bf16 B[K,N] row-major
//      3 = bf16 B[k][n]=Bg[n*ldb+k]
// EPI: 0 = err  (acc + bias[col] - target) -> bf16 out + per-block col-sums
//      1 = gW   W_new = W - lr_w * coef*acc -> bf16 out
//      2 = out  acc + bias[col] + addf -> f32 out
// ---------------------------------------------------------------------------
template<int AM, int BMO, int EPI>
__global__ __launch_bounds__(NT)
void gemm_kernel(const void* __restrict__ Ap, const void* __restrict__ Bp,
                 int K, int lda, int ldb, int ldc,
                 const float* __restrict__ bias,
                 const unsigned short* __restrict__ tgt,
                 const float* __restrict__ addf,
                 const float* __restrict__ Wf,
                 const float* __restrict__ lrwp,
                 float coef,
                 unsigned short* __restrict__ obf,
                 float* __restrict__ of,
                 float* __restrict__ gbp)
{
  __shared__ __align__(16) unsigned short As[TMg * LDK];
  __shared__ __align__(16) unsigned short Bs[TNg * LDK];
  __shared__ float colsum[TNg];

  const int tid = threadIdx.x, wave = tid >> 5, lane = tid & 31;
  const int lh = lane >> 4, li = lane & 15;
  const int wm = (wave >> 1) * 32, wn = (wave & 1) * 64;
  const int m0 = blockIdx.y * TMg, n0 = blockIdx.x * TNg;

  if (EPI == 0 && tid < TNg) colsum[tid] = 0.0f;

  const v8f vz = {0.f,0.f,0.f,0.f,0.f,0.f,0.f,0.f};
  v8f acc[2][4];
  for (int i = 0; i < 2; i++)
    for (int j = 0; j < 4; j++) acc[i][j] = vz;

  for (int k0 = 0; k0 < K; k0 += TK) {
    if constexpr (AM == 0) {
      // bf16 [M,K], k-contiguous: b128 load -> b128 LDS store
      const unsigned short* Ag = (const unsigned short*)Ap;
      for (int q = tid; q < (TMg * TK) / 8; q += NT) {   // 256 chunks of 8
        int m = q >> 2, kc = (q & 3) * 8;
        *reinterpret_cast<uint4*>(&As[m * LDK + kc]) =
            *reinterpret_cast<const uint4*>(Ag + (size_t)(m0 + m) * lda + (k0 + kc));
      }
    } else {
      // bf16 [K,M], m-contiguous: b128 load, scatter-store to As[m][k]
      const unsigned short* Ag = (const unsigned short*)Ap;
      for (int q = tid; q < (TMg * TK) / 8; q += NT) {   // 256 chunks of 8
        int k = q >> 3, mc = (q & 7) * 8;
        union { uint4 u; unsigned short s[8]; } t;
        t.u = *reinterpret_cast<const uint4*>(Ag + (size_t)(k0 + k) * lda + (m0 + mc));
        #pragma unroll
        for (int j = 0; j < 8; j++) As[(mc + j) * LDK + k] = t.s[j];
      }
    }
    if constexpr (BMO == 1) {
      // f32 [N,K], k-contiguous: float4 load -> packed uint2 LDS store
      const float* Bg = (const float*)Bp;
      for (int q = tid; q < (TNg * TK) / 4; q += NT) {   // 1024 chunks of 4
        int n = q >> 3, kc = (q & 7) * 4;
        float4 f = *reinterpret_cast<const float4*>(Bg + (size_t)(n0 + n) * ldb + (k0 + kc));
        uint2 r; r.x = pk2(f.x, f.y); r.y = pk2(f.z, f.w);
        *reinterpret_cast<uint2*>(&Bs[n * LDK + kc]) = r;
      }
    } else if constexpr (BMO == 2) {
      // bf16 [K,N], n-contiguous: b128 load, scatter-store to Bs[n][k]
      const unsigned short* Bg = (const unsigned short*)Bp;
      for (int q = tid; q < (TNg * TK) / 8; q += NT) {   // 512 chunks of 8
        int k = q >> 4, nc = (q & 15) * 8;
        union { uint4 u; unsigned short s[8]; } t;
        t.u = *reinterpret_cast<const uint4*>(Bg + (size_t)(k0 + k) * ldb + (n0 + nc));
        #pragma unroll
        for (int j = 0; j < 8; j++) Bs[(nc + j) * LDK + k] = t.s[j];
      }
    } else {
      // bf16 [N,K], k-contiguous: b128 load -> b128 LDS store
      const unsigned short* Bg = (const unsigned short*)Bp;
      for (int q = tid; q < (TNg * TK) / 8; q += NT) {   // 512 chunks of 8
        int n = q >> 2, kc = (q & 3) * 8;
        *reinterpret_cast<uint4*>(&Bs[n * LDK + kc]) =
            *reinterpret_cast<const uint4*>(Bg + (size_t)(n0 + n) * ldb + (k0 + kc));
      }
    }
    __syncthreads();
    v16bf af0 = ldfrag(&As[(wm + li) * LDK], lh);
    v16bf af1 = ldfrag(&As[(wm + 16 + li) * LDK], lh);
    #pragma unroll
    for (int nf = 0; nf < 4; nf++) {
      v16bf bf = ldfrag(&Bs[(wn + nf * 16 + li) * LDK], lh);
      acc[0][nf] = __builtin_amdgcn_wmma_f32_16x16x32_bf16(false, af0, false, bf, (short)0, acc[0][nf], false, false);
      acc[1][nf] = __builtin_amdgcn_wmma_f32_16x16x32_bf16(false, af1, false, bf, (short)0, acc[1][nf], false, false);
    }
    __syncthreads();
  }

  for (int mf = 0; mf < 2; mf++)
    for (int nf = 0; nf < 4; nf++) {
      float cs = 0.0f;
      for (int r = 0; r < 8; r++) {
        int row = m0 + wm + mf * 16 + lh * 8 + r;
        int col = n0 + wn + nf * 16 + li;
        size_t o = (size_t)row * ldc + col;
        float v = acc[mf][nf][r];
        if constexpr (EPI == 0) {
          float e = v + bias[col] - bf2f(tgt[o]);
          obf[o] = f2bf(e);
          cs += e;
        } else if constexpr (EPI == 1) {
          obf[o] = f2bf(Wf[o] - lrwp[o] * (coef * v));
        } else {
          of[o] = v + bias[col] + addf[o];
        }
      }
      if constexpr (EPI == 0)
        atomicAdd(&colsum[wn + nf * 16 + li], cs);   // ds_add_f32
    }
  if constexpr (EPI == 0) {
    __syncthreads();
    if (tid < TNg) gbp[(size_t)blockIdx.y * Dd + (n0 + tid)] = colsum[tid];
  }
}

// ---------------------------------------------------------------------------
// Kernel: b_new = b - lr_b * coef * sum_m gbpart[m][i]
// ---------------------------------------------------------------------------
__global__ void bnew_kernel(const float* __restrict__ b,
                            const float* __restrict__ lrb,
                            const float* __restrict__ gbp,
                            float* __restrict__ bnew,
                            float coef, int nparts)
{
  int i = blockIdx.x * blockDim.x + threadIdx.x;
  if (i < Dd) {
    float s = 0.0f;
    for (int m = 0; m < nparts; m++) s += gbp[(size_t)m * Dd + i];
    bnew[i] = b[i] - lrb[i] * (coef * s);
  }
}

// ---------------------------------------------------------------------------
extern "C" void kernel_launch(void* const* d_in, const int* in_sizes, int n_in,
                              void* d_out, int out_size, void* d_ws, size_t ws_size,
                              hipStream_t stream) {
  (void)in_sizes; (void)n_in; (void)out_size; (void)ws_size;
  const float* src = (const float*)d_in[0];
  const float* thk = (const float*)d_in[1];
  const float* thq = (const float*)d_in[2];
  const float* thv = (const float*)d_in[3];
  const float* W   = (const float*)d_in[4];
  const float* bb  = (const float*)d_in[5];
  const float* lrw = (const float*)d_in[6];
  const float* lrb = (const float*)d_in[7];
  float* out = (float*)d_out;

  char* p = (char*)d_ws;
  size_t off = 0;
  auto take = [&](size_t bytes) { char* r = p + off; off = (off + bytes + 255) & ~(size_t)255; return (void*)r; };
  const size_t BD = (size_t)Bt * Dd;
  const size_t DDsz = (size_t)Dd * Dd;
  unsigned short* trainb  = (unsigned short*)take(BD * 2);
  unsigned short* testb   = (unsigned short*)take(BD * 2);
  float*          testf   = (float*)take(BD * 4);
  unsigned short* targetb = (unsigned short*)take(BD * 2);
  unsigned short* errb    = (unsigned short*)take(BD * 2);
  unsigned short* wnewb   = (unsigned short*)take(DDsz * 2);
  float*          gbpart  = (float*)take((size_t)(Bt / TMg) * Dd * 4);
  float*          bnew    = (float*)take((size_t)Dd * 4);

  const float coef = 2.0f / (float)(Bt * (size_t)Dd);   // 2 / (B*D)
  dim3 blk(NT);
  dim3 gP(Dd / TNp, Bt / TMp);    // (32, 64) proj over [B, D]
  dim3 gBD(Dd / TNg, Bt / TMg);   // (16, 64) for [B, D] outputs
  dim3 gDD(Dd / TNg, Dd / TMg);   // (16, 32) for [D, D] outputs

  // 1) projections + target
  proj_kernel<<<gP, blk, 0, stream>>>(src, thk, thq, thv, trainb, testb, testf, targetb);
  // 2) err = train@W.T + b - target  (+ per-block column sums -> gbpart)
  gemm_kernel<0,1,0><<<gBD, blk, 0, stream>>>(trainb, W, Dd, Dd, Dd, Dd,
      bb, targetb, nullptr, nullptr, nullptr, coef, errb, nullptr, gbpart);
  // 3) W_new = W - lr_w * coef*(err.T @ train)   (K = 4096)
  gemm_kernel<1,2,1><<<gDD, blk, 0, stream>>>(errb, trainb, Bt, Dd, Dd, Dd,
      nullptr, nullptr, nullptr, W, lrw, coef, wnewb, nullptr, nullptr);
  // 4) b_new = b - lr_b * coef * gb
  bnew_kernel<<<dim3(Dd / 256), dim3(256), 0, stream>>>(bb, lrb, gbpart, bnew, coef, Bt / TMg);
  // 5) z = test @ W_new.T + b_new + test
  gemm_kernel<0,3,2><<<gBD, blk, 0, stream>>>(testb, wnewb, Dd, Dd, Dd, Dd,
      bnew, nullptr, testf, nullptr, nullptr, coef, nullptr, out, nullptr);
}
